// TransformerDecoderLayer_76244259439043
// MI455X (gfx1250) — compile-verified, matrix operand order checked by reading, not run on hardware
//
#include <hip/hip_runtime.h>
#include <hip/hip_bf16.h>

typedef __bf16 bf16_t;
typedef __attribute__((ext_vector_type(16))) __bf16 v16bf;
typedef __attribute__((ext_vector_type(8)))  __bf16 v8bf;
typedef __attribute__((ext_vector_type(8)))  float  v8f;

// Plain 16B global->LDS copy (vector load + ds_store_b128). The gfx1250
// async-to-LDS builtin exists but takes pointers in clang's internal
// __device__/__shared__ language address spaces that HIP source cannot
// spell, so we keep the synchronous path.
__device__ __forceinline__ void cp16_g2l(const bf16_t* g, bf16_t* l) {
    *(v8bf*)l = *(const v8bf*)g;
}

__device__ __forceinline__ v8bf zero8() {
    v8bf v;
#pragma unroll
    for (int j = 0; j < 8; ++j) v[j] = (bf16_t)0.0f;
    return v;
}

// ---------------------------------------------------------------------------
// Batch offset helper: off(z) = (z%m0)*s0 + ((z/m0)%m1)*s1 + (z/(m0*m1))*s2
// ---------------------------------------------------------------------------
__device__ __forceinline__ long zoff(long z, long s0, long s1, long s2, int m0, int m1) {
    long z0 = z % m0;
    long t  = z / m0;
    long z1 = t % m1;
    long z2 = t / m1;
    return z0 * s0 + z1 * s1 + z2 * s2;
}

// ---------------------------------------------------------------------------
// Generic batched bf16 WMMA GEMM:  C = act(alpha * A*B + bias)
// A: [M,K] row stride lda (bf16).  B: [K,N] stride ldb, or [N,K] if TRANSB.
// Requirements (guaranteed by caller): K % 8 == 0, N % 8 == 0, all base
// offsets and strides multiples of 8 elements (16B alignment for v8bf ops).
//
// Block: 128 threads (4 waves). Tile: 128(M) x 64(N), K-step 32.
// Wave w computes rows [32w, 32w+32) x 64 cols = 8 WMMA per K-step.
// LDS: A tile row-major [m][k]; B tile n-major [n][k] so each lane's fragment
// is two contiguous 16B chunks -> ds_load_b128 pairs (ISA 7.12.2 layout).
// Full tiles take an unguarded staging fast path (no exec divergence).
// ---------------------------------------------------------------------------
#define LDSPAD 40   // row stride in elements (80B): disjoint bank groups, 16B aligned

template <bool TRANSB>
__global__ __launch_bounds__(128) void k_bgemm(
    const bf16_t* __restrict__ A, const bf16_t* __restrict__ B,
    float* __restrict__ Cf, bf16_t* __restrict__ Cb,
    int M, int N, int K, int lda, int ldb, int ldc,
    long as0, long as1, long as2, int am0, int am1,
    long bs0, long bs1, long bs2, int bm0, int bm1,
    long cs0, long cs1, long cs2, int cm0, int cm1,
    const float* __restrict__ bias, int act, float alpha)
{
    __shared__ bf16_t As[128][LDSPAD];   // [m][k]
    __shared__ bf16_t Bs[64][LDSPAD];    // [n][k]

    const int  tid   = threadIdx.x;
    const int  mBase = blockIdx.y * 128;
    const int  nBase = blockIdx.x * 64;
    const long z     = blockIdx.z;

    const bf16_t* Ab = A + zoff(z, as0, as1, as2, am0, am1);
    const bf16_t* Bb = B + zoff(z, bs0, bs1, bs2, bm0, bm1);
    const long coff  = zoff(z, cs0, cs1, cs2, cm0, cm1);

    v8f acc[2][4] = {};

    const int wid  = tid >> 5;
    const int lane = tid & 31;
    const int half = lane >> 4;   // 0: lanes 0-15, 1: lanes 16-31
    const int l15  = lane & 15;

    const bool fullM = (mBase + 128 <= M);
    const bool fullN = (nBase + 64 <= N);

    for (int k0 = 0; k0 < K; k0 += 32) {
        const bool fullK = (k0 + 32 <= K);

        // ---- Stage A tile: 128 rows x 32 k, as 512 chunks of 8 (16B)
        if (fullM && fullK) {
#pragma unroll
            for (int s = 0; s < 4; ++s) {
                int c   = tid + s * 128;
                int r   = c >> 2;
                int col = (c & 3) * 8;
                cp16_g2l(Ab + (long)(mBase + r) * lda + (k0 + col), &As[r][col]);
            }
        } else {
#pragma unroll
            for (int s = 0; s < 4; ++s) {
                int c   = tid + s * 128;
                int r   = c >> 2;
                int col = (c & 3) * 8;
                int gm  = mBase + r, gk = k0 + col;
                v8bf v  = zero8();
                if (gm < M && gk < K) v = *(const v8bf*)(Ab + (long)gm * lda + gk);
                *(v8bf*)(&As[r][col]) = v;
            }
        }
        // ---- Stage B tile into [n][k]
        if (TRANSB) {
            if (fullN && fullK) {
#pragma unroll
                for (int s = 0; s < 2; ++s) {
                    int c   = tid + s * 128;
                    int nn  = c >> 2;
                    int col = (c & 3) * 8;
                    cp16_g2l(Bb + (long)(nBase + nn) * ldb + (k0 + col), &Bs[nn][col]);
                }
            } else {
#pragma unroll
                for (int s = 0; s < 2; ++s) {
                    int c   = tid + s * 128;
                    int nn  = c >> 2;
                    int col = (c & 3) * 8;
                    int gn  = nBase + nn, gk = k0 + col;
                    v8bf v  = zero8();
                    if (gn < N && gk < K) v = *(const v8bf*)(Bb + (long)gn * ldb + gk);
                    *(v8bf*)(&Bs[nn][col]) = v;
                }
            }
        } else {
            // B stored [K,N]: load 8 n-contiguous, scatter-transpose into LDS.
#pragma unroll
            for (int s = 0; s < 2; ++s) {
                int c  = tid + s * 128;
                int kk = c >> 3;
                int nn = (c & 7) * 8;
                int gk = k0 + kk, gn = nBase + nn;
                v8bf v = zero8();
                if (gk < K && gn < N) v = *(const v8bf*)(Bb + (long)gk * ldb + gn);
#pragma unroll
                for (int j = 0; j < 8; ++j) Bs[nn + j][kk] = v[j];
            }
        }
        __syncthreads();

        // ---- Fragments: two ds_load_b128 each (ISA 7.12.2 16-bit layout)
        v16bf af[2];
#pragma unroll
        for (int mt = 0; mt < 2; ++mt) {
            const bf16_t* rp = &As[wid * 32 + mt * 16 + l15][0];
            v8bf a0 = *(const v8bf*)(rp + half * 8);
            v8bf a1 = *(const v8bf*)(rp + 16 + half * 8);
            af[mt]  = __builtin_shufflevector(a0, a1, 0, 1, 2, 3, 4, 5, 6, 7,
                                              8, 9, 10, 11, 12, 13, 14, 15);
        }
#pragma unroll
        for (int nt = 0; nt < 4; ++nt) {
            const bf16_t* bp = &Bs[nt * 16 + l15][0];
            v8bf b0 = *(const v8bf*)(bp + half * 8);
            v8bf b1 = *(const v8bf*)(bp + 16 + half * 8);
            v16bf bfr = __builtin_shufflevector(b0, b1, 0, 1, 2, 3, 4, 5, 6, 7,
                                                8, 9, 10, 11, 12, 13, 14, 15);
#pragma unroll
            for (int mt = 0; mt < 2; ++mt)
                acc[mt][nt] = __builtin_amdgcn_wmma_f32_16x16x32_bf16(
                    false, af[mt], false, bfr, (short)0, acc[mt][nt], false, false);
        }
        __syncthreads();
    }

    // ---- Epilogue: VGPR i -> M = base + half*8 + i, N = nBase + nt*16 + l15
    const bool fullC = fullM && fullN;
#pragma unroll
    for (int mt = 0; mt < 2; ++mt) {
#pragma unroll
        for (int nt = 0; nt < 4; ++nt) {
            int gn = nBase + nt * 16 + l15;
#pragma unroll
            for (int i = 0; i < 8; ++i) {
                int gm = mBase + wid * 32 + mt * 16 + half * 8 + i;
                if (fullC || (gm < M && gn < N)) {
                    float v = acc[mt][nt][i] * alpha;
                    if (bias) v += bias[gn];
                    if (act == 1)      v = fmaxf(v, 0.0f);
                    else if (act == 2) v = 1.0f / (1.0f + __expf(-v));
                    long idx = coff + (long)gm * ldc + gn;
                    if (Cf) Cf[idx] = v;
                    if (Cb) Cb[idx] = (bf16_t)v;
                }
            }
        }
    }
}

// ---------------------------------------------------------------------------
// f32 -> bf16 convert
// ---------------------------------------------------------------------------
__global__ __launch_bounds__(256) void k_cvt(const float* __restrict__ s,
                                             bf16_t* __restrict__ d, long n) {
    long i = (long)blockIdx.x * 256 + threadIdx.x;
    if (i < n) d[i] = (bf16_t)s[i];
}

// f32 [R,Cc] -> bf16 transposed [Cc,R]
__global__ __launch_bounds__(256) void k_cvt_t(const float* __restrict__ s,
                                               bf16_t* __restrict__ d, int R, int Cc) {
    long i = (long)blockIdx.x * 256 + threadIdx.x;
    long n = (long)R * Cc;
    if (i < n) {
        long r = i / Cc;
        long c = i % Cc;
        d[c * R + r] = (bf16_t)s[i];
    }
}

// a + b -> bf16
__global__ __launch_bounds__(256) void k_add_bf16(const float* __restrict__ a,
                                                  const float* __restrict__ b,
                                                  bf16_t* __restrict__ d, long n) {
    long i = (long)blockIdx.x * 256 + threadIdx.x;
    if (i < n) d[i] = (bf16_t)(a[i] + b[i]);
}

// concat [row,256]+[row,256] -> bf16 [row,512]
__global__ __launch_bounds__(256) void k_concat(const float* __restrict__ a,
                                                const float* __restrict__ b,
                                                bf16_t* __restrict__ d, long n) {
    long i = (long)blockIdx.x * 256 + threadIdx.x;
    if (i < n) {
        long row = i >> 9;
        int  c   = (int)(i & 511);
        float v  = (c < 256) ? a[row * 256 + c] : b[row * 256 + (c - 256)];
        d[i] = (bf16_t)v;
    }
}

// ---------------------------------------------------------------------------
// Row softmax over 200 columns, f32 in -> bf16 probs out. One block per row.
// ---------------------------------------------------------------------------
__global__ __launch_bounds__(256) void k_softmax200(const float* __restrict__ in,
                                                    bf16_t* __restrict__ out) {
    long row = blockIdx.x;
    const float* x = in + row * 200;
    bf16_t*      o = out + row * 200;
    int tid = threadIdx.x;
    __shared__ float red[256];

    float v = (tid < 200) ? x[tid] : -3.4e38f;
    red[tid] = v;
    __syncthreads();
    for (int s = 128; s > 0; s >>= 1) {
        if (tid < s) red[tid] = fmaxf(red[tid], red[tid + s]);
        __syncthreads();
    }
    float mx = red[0];
    __syncthreads();

    float e = (tid < 200) ? __expf(v - mx) : 0.0f;
    red[tid] = e;
    __syncthreads();
    for (int s = 128; s > 0; s >>= 1) {
        if (tid < s) red[tid] += red[tid + s];
        __syncthreads();
    }
    float inv = 1.0f / red[0];
    if (tid < 200) o[tid] = (bf16_t)(e * inv);
}

// ---------------------------------------------------------------------------
// Fused residual / gate / clip + LayerNorm over C=256. One block (256 thr)/row.
// mode 0: x=a+d | mode 1: x=gv[:,c]*a + gv[:,256+c]*d | mode 2: x=clip(a+d)
// ---------------------------------------------------------------------------
__global__ __launch_bounds__(256) void k_ln(const float* __restrict__ a,
                                            const float* __restrict__ d,
                                            const float* __restrict__ gv,
                                            const float* __restrict__ gamma,
                                            const float* __restrict__ beta,
                                            float* __restrict__ outf,
                                            bf16_t* __restrict__ outb, int mode) {
    long row = blockIdx.x;
    int  c   = threadIdx.x;
    float av = a[row * 256 + c];
    float dv = d[row * 256 + c];
    float x;
    if (mode == 0) {
        x = av + dv;
    } else if (mode == 1) {
        float g1 = gv[row * 512 + c];
        float g2 = gv[row * 512 + 256 + c];
        x = g1 * av + g2 * dv;
    } else {
        x = av + dv;
        x = fminf(fmaxf(x, -65504.0f), 65504.0f);
    }
    __shared__ float red[256];
    red[c] = x;
    __syncthreads();
    for (int s = 128; s > 0; s >>= 1) {
        if (c < s) red[c] += red[c + s];
        __syncthreads();
    }
    float mu = red[0] * (1.0f / 256.0f);
    __syncthreads();
    float xm = x - mu;
    red[c] = xm * xm;
    __syncthreads();
    for (int s = 128; s > 0; s >>= 1) {
        if (c < s) red[c] += red[c + s];
        __syncthreads();
    }
    float var = red[0] * (1.0f / 256.0f);
    float y   = xm * rsqrtf(var + 1e-5f) * gamma[c] + beta[c];
    if (outf) outf[row * 256 + c] = y;
    if (outb) outb[row * 256 + c] = (bf16_t)y;
}

// ---------------------------------------------------------------------------
// Block-2 attention: sequences of P=8 points per (batch, query), 8 heads.
// One 64-thread block per (bi, qi); thread = (h, p). QKV from qkp/vp (bf16).
// ---------------------------------------------------------------------------
__global__ __launch_bounds__(64) void k_attn8(const bf16_t* __restrict__ qkp,
                                              const bf16_t* __restrict__ vp,
                                              bf16_t* __restrict__ outb) {
    __shared__ float Qs[8][256];
    __shared__ float Ks[8][256];
    __shared__ float Vs[8][256];
    int  g    = blockIdx.x;         // 0..1599 = bi*200 + qi
    int  bi   = g / 200;
    int  qi   = g % 200;
    long tok0 = (long)bi * 1600 + (long)qi * 8;
    int  tid  = threadIdx.x;

    for (int i = tid; i < 8 * 256; i += 64) {
        int  p = i >> 8, c = i & 255;
        long t = tok0 + p;
        Qs[p][c] = (float)qkp[t * 512 + c];
        Ks[p][c] = (float)qkp[t * 512 + 256 + c];
        Vs[p][c] = (float)vp[t * 256 + c];
    }
    __syncthreads();

    int p = tid & 7, h = tid >> 3;
    float s[8];
#pragma unroll
    for (int k = 0; k < 8; ++k) {
        float acc = 0.0f;
        for (int dd = 0; dd < 32; ++dd) acc += Qs[p][h * 32 + dd] * Ks[k][h * 32 + dd];
        s[k] = acc * 0.17677669529663687f;   // 1/sqrt(32)
    }
    float mx = s[0];
    for (int k = 1; k < 8; ++k) mx = fmaxf(mx, s[k]);
    float sum = 0.0f;
    for (int k = 0; k < 8; ++k) { s[k] = __expf(s[k] - mx); sum += s[k]; }
    float inv = 1.0f / sum;
    for (int dd = 0; dd < 32; ++dd) {
        float o = 0.0f;
        for (int k = 0; k < 8; ++k) o += s[k] * Vs[k][h * 32 + dd];
        outb[(tok0 + p) * 256 + h * 32 + dd] = (bf16_t)(o * inv);
    }
}

// ---------------------------------------------------------------------------
// Multi-scale deformable sampling. One 256-thread block per token:
// thread = (h, d). 16 points, 4 levels, bilinear zero-pad sampling.
// ---------------------------------------------------------------------------
__global__ __launch_bounds__(256) void k_deform(const float* __restrict__ so,
                                                const float* __restrict__ aw,
                                                const float* __restrict__ rp,
                                                const float* __restrict__ value,
                                                float* __restrict__ outp) {
    long tok = blockIdx.x;
    int  bi  = (int)(tok / 1600);
    int  h   = threadIdx.x >> 5;
    int  d   = threadIdx.x & 31;

    float cx = rp[tok * 4 + 0], cy = rp[tok * 4 + 1];
    float rw = rp[tok * 4 + 2], rh = rp[tok * 4 + 3];
    const float* ofs = so + tok * 256 + h * 32;   // [16][2]
    const float* lg  = aw + tok * 128 + h * 16;   // [16]

    float w[16];
    float mx = lg[0];
    for (int p = 1; p < 16; ++p) mx = fmaxf(mx, lg[p]);
    float sum = 0.0f;
    for (int p = 0; p < 16; ++p) { w[p] = __expf(lg[p] - mx); sum += w[p]; }
    float invs = 1.0f / sum;

    const int dims[4]   = {80, 40, 20, 10};
    const int starts[4] = {0, 6400, 8000, 8400};
    float acc = 0.0f;
    for (int p = 0; p < 16; ++p) {
        int   lvl = p >> 2;
        int   W = dims[lvl], H = dims[lvl];
        float lx = cx + ofs[2 * p + 0] * 0.25f * rw * 0.5f;
        float ly = cy + ofs[2 * p + 1] * 0.25f * rh * 0.5f;
        float x  = lx * (float)W - 0.5f;
        float y  = ly * (float)H - 0.5f;
        float xf = floorf(x), yf = floorf(y);
        int   x0 = (int)xf, y0 = (int)yf;
        float wx1 = x - xf, wy1 = y - yf, wx0 = 1.0f - wx1, wy0 = 1.0f - wy1;
        const float* vb = value + ((long)bi * 8500 + starts[lvl]) * 256 + h * 32 + d;
        float s00 = 0, s10 = 0, s01 = 0, s11 = 0;
        if (x0 >= 0 && x0 < W && y0 >= 0 && y0 < H)         s00 = vb[(long)(y0 * W + x0) * 256];
        if (x0 + 1 >= 0 && x0 + 1 < W && y0 >= 0 && y0 < H) s10 = vb[(long)(y0 * W + x0 + 1) * 256];
        if (x0 >= 0 && x0 < W && y0 + 1 >= 0 && y0 + 1 < H) s01 = vb[(long)((y0 + 1) * W + x0) * 256];
        if (x0 + 1 >= 0 && x0 + 1 < W && y0 + 1 >= 0 && y0 + 1 < H)
                                                            s11 = vb[(long)((y0 + 1) * W + x0 + 1) * 256];
        float sv = s00 * wx0 * wy0 + s10 * wx1 * wy0 + s01 * wx0 * wy1 + s11 * wx1 * wy1;
        acc += sv * (w[p] * invs);
    }
    outp[tok * 256 + h * 32 + d] = acc;
}

// ---------------------------------------------------------------------------
// Host orchestration
// ---------------------------------------------------------------------------
extern "C" void kernel_launch(void* const* d_in, const int* in_sizes, int n_in,
                              void* d_out, int out_size, void* d_ws, size_t ws_size,
                              hipStream_t stream) {
    (void)in_sizes; (void)n_in; (void)out_size; (void)ws_size;
    const float* target = (const float*)d_in[0];
    const float* refp   = (const float*)d_in[1];
    const float* value  = (const float*)d_in[2];
    const float* pe     = (const float*)d_in[4];
    const float* inw    = (const float*)d_in[5];
    const float* inb    = (const float*)d_in[6];
    const float* opw    = (const float*)d_in[7];
    const float* opb    = (const float*)d_in[8];
    const float* sow    = (const float*)d_in[9];
    const float* sob    = (const float*)d_in[10];
    const float* aww    = (const float*)d_in[11];
    const float* awb    = (const float*)d_in[12];
    const float* gw     = (const float*)d_in[13];
    const float* gb     = (const float*)d_in[14];
    const float* gng    = (const float*)d_in[15];
    const float* gnb    = (const float*)d_in[16];
    const float* n1g    = (const float*)d_in[17];
    const float* n1b    = (const float*)d_in[18];
    const float* n2g    = (const float*)d_in[19];
    const float* n2b    = (const float*)d_in[20];
    const float* n3g    = (const float*)d_in[21];
    const float* n3b    = (const float*)d_in[22];
    const float* w1     = (const float*)d_in[23];
    const float* b1     = (const float*)d_in[24];
    const float* w2     = (const float*)d_in[25];
    const float* b2     = (const float*)d_in[26];
    float* out = (float*)d_out;

    const long BN = 12800;   // 8 * 1600 tokens

    char*  wsp = (char*)d_ws;
    size_t off = 0;
    auto alloc = [&](size_t bytes) -> void* {
        void* p = wsp + off;
        off += (bytes + 255) & ~(size_t)255;
        return p;
    };

    // bf16 transposed weights: W^T [N,K]
    bf16_t* inw_t  = (bf16_t*)alloc(768L * 256 * 2);
    bf16_t* opw_t  = (bf16_t*)alloc(256L * 256 * 2);
    bf16_t* sow_t  = (bf16_t*)alloc(256L * 256 * 2);
    bf16_t* aww_t  = (bf16_t*)alloc(128L * 256 * 2);
    bf16_t* gw_t   = (bf16_t*)alloc(512L * 512 * 2);
    bf16_t* w1_t   = (bf16_t*)alloc(1024L * 256 * 2);
    bf16_t* w2_t   = (bf16_t*)alloc(256L * 1024 * 2);
    // activations
    bf16_t* tA     = (bf16_t*)alloc(BN * 256 * 2);
    bf16_t* qb     = (bf16_t*)alloc(BN * 256 * 2);
    bf16_t* qkp    = (bf16_t*)alloc(BN * 512 * 2);
    bf16_t* vp     = (bf16_t*)alloc(BN * 256 * 2);
    float*  scores = (float*)alloc(512L * 200 * 200 * 4);
    bf16_t* probs  = (bf16_t*)alloc(512L * 200 * 200 * 2);
    bf16_t* attnA  = (bf16_t*)alloc(BN * 256 * 2);
    float*  mhao   = (float*)alloc(BN * 256 * 4);
    float*  t1f    = (float*)alloc(BN * 256 * 4);
    bf16_t* t1b    = (bf16_t*)alloc(BN * 256 * 2);
    float*  t2f    = (float*)alloc(BN * 256 * 4);
    bf16_t* t2b    = (bf16_t*)alloc(BN * 256 * 2);
    float*  sof    = (float*)alloc(BN * 256 * 4);
    float*  awf    = (float*)alloc(BN * 128 * 4);
    float*  t2d    = (float*)alloc(BN * 256 * 4);
    bf16_t* gi     = (bf16_t*)alloc(BN * 512 * 2);
    float*  gatev  = (float*)alloc(BN * 512 * 4);
    float*  t3f    = (float*)alloc(BN * 256 * 4);
    bf16_t* t3b    = (bf16_t*)alloc(BN * 256 * 2);
    bf16_t* h1     = (bf16_t*)alloc(BN * 1024 * 2);
    float*  ffn    = (float*)alloc(BN * 256 * 4);

    auto cvtT = [&](const float* s, bf16_t* d, int R, int Cc) {
        long n = (long)R * Cc;
        k_cvt_t<<<dim3((unsigned)((n + 255) / 256)), dim3(256), 0, stream>>>(s, d, R, Cc);
    };
    auto cvt = [&](const float* s, bf16_t* d, long n) {
        k_cvt<<<dim3((unsigned)((n + 255) / 256)), dim3(256), 0, stream>>>(s, d, n);
    };
    auto gemm = [&](bool transB, const bf16_t* A, const bf16_t* B, float* Cf, bf16_t* Cb,
                    int M, int N, int K, int lda, int ldb, int ldc, int batch,
                    long as0, long as1, long as2, int am0, int am1,
                    long bs0, long bs1, long bs2, int bm0, int bm1,
                    long cs0, long cs1, long cs2, int cm0, int cm1,
                    const float* bias, int act, float alpha) {
        dim3 g((N + 63) / 64, (M + 127) / 128, batch);
        if (transB)
            k_bgemm<true><<<g, dim3(128), 0, stream>>>(A, B, Cf, Cb, M, N, K, lda, ldb, ldc,
                as0, as1, as2, am0, am1, bs0, bs1, bs2, bm0, bm1,
                cs0, cs1, cs2, cm0, cm1, bias, act, alpha);
        else
            k_bgemm<false><<<g, dim3(128), 0, stream>>>(A, B, Cf, Cb, M, N, K, lda, ldb, ldc,
                as0, as1, as2, am0, am1, bs0, bs1, bs2, bm0, bm1,
                cs0, cs1, cs2, cm0, cm1, bias, act, alpha);
    };

    // weight conversions (transposed: rows become N dim of GEMM)
    cvtT(inw, inw_t, 256, 768);    // [768,256]
    cvtT(opw, opw_t, 256, 256);
    cvtT(sow, sow_t, 256, 256);
    cvtT(aww, aww_t, 256, 128);    // [128,256]
    cvtT(gw,  gw_t,  512, 512);
    cvtT(w1,  w1_t,  256, 1024);   // [1024,256]
    cvtT(w2,  w2_t,  1024, 256);   // [256,1024]

    // ---- Block 1: attention across queries (seq len 200, 64 groups) ----
    cvt(target, tA, BN * 256);
    k_add_bf16<<<dim3((unsigned)((BN * 256 + 255) / 256)), dim3(256), 0, stream>>>(target, pe, qb, BN * 256);

    // Q|K projection (rows 0..511 of inw_t), V projection (rows 512..767)
    gemm(true, qb, inw_t, nullptr, qkp, (int)BN, 512, 256, 256, 256, 512, 1,
         0,0,0,1,1, 0,0,0,1,1, 0,0,0,1,1, inb, 0, 1.0f);
    gemm(true, tA, inw_t + 512L * 256, nullptr, vp, (int)BN, 256, 256, 256, 256, 256, 1,
         0,0,0,1,1, 0,0,0,1,1, 0,0,0,1,1, inb + 512, 0, 1.0f);

    // scores[z=bi*64+p*8+h][200][200] = Q * K^T / sqrt(32)
    gemm(true, qkp, qkp + 256, scores, nullptr, 200, 200, 32, 4096, 4096, 200, 512,
         32, 512, 819200, 8, 8,       // A offsets: h,p,bi over qkp
         32, 512, 819200, 8, 8,       // B offsets (K part, base ptr pre-offset +256)
         40000, 0, 0, 512, 1,         // C: z*40000
         nullptr, 0, 0.17677669529663687f);
    k_softmax200<<<dim3(102400), dim3(256), 0, stream>>>(scores, probs);

    // attn @ V  -> head-interleaved bf16 token matrix
    gemm(false, probs, vp, nullptr, attnA, 200, 32, 200, 200, 2048, 2048, 512,
         40000, 0, 0, 512, 1,         // A: z*40000
         32, 256, 409600, 8, 8,       // B: h,p,bi over vp
         32, 256, 409600, 8, 8,       // C: h,p,bi over attnA
         nullptr, 0, 1.0f);

    // out_proj + residual + LN1
    gemm(true, attnA, opw_t, mhao, nullptr, (int)BN, 256, 256, 256, 256, 256, 1,
         0,0,0,1,1, 0,0,0,1,1, 0,0,0,1,1, opb, 0, 1.0f);
    k_ln<<<dim3((unsigned)BN), dim3(256), 0, stream>>>(target, mhao, nullptr, n1g, n1b, t1f, t1b, 0);

    // ---- Block 2: attention across points (seq len 8) ----
    k_add_bf16<<<dim3((unsigned)((BN * 256 + 255) / 256)), dim3(256), 0, stream>>>(t1f, pe, qb, BN * 256);
    gemm(true, qb, inw_t, nullptr, qkp, (int)BN, 512, 256, 256, 256, 512, 1,
         0,0,0,1,1, 0,0,0,1,1, 0,0,0,1,1, inb, 0, 1.0f);
    gemm(true, t1b, inw_t + 512L * 256, nullptr, vp, (int)BN, 256, 256, 256, 256, 256, 1,
         0,0,0,1,1, 0,0,0,1,1, 0,0,0,1,1, inb + 512, 0, 1.0f);
    k_attn8<<<dim3(1600), dim3(64), 0, stream>>>(qkp, vp, attnA);
    gemm(true, attnA, opw_t, mhao, nullptr, (int)BN, 256, 256, 256, 256, 256, 1,
         0,0,0,1,1, 0,0,0,1,1, 0,0,0,1,1, opb, 0, 1.0f);
    k_ln<<<dim3((unsigned)BN), dim3(256), 0, stream>>>(t1f, mhao, nullptr, n2g, n2b, t2f, t2b, 0);

    // ---- Deformable cross-attention ----
    k_add_bf16<<<dim3((unsigned)((BN * 256 + 255) / 256)), dim3(256), 0, stream>>>(t2f, pe, qb, BN * 256);
    gemm(true, qb, sow_t, sof, nullptr, (int)BN, 256, 256, 256, 256, 256, 1,
         0,0,0,1,1, 0,0,0,1,1, 0,0,0,1,1, sob, 0, 1.0f);
    gemm(true, qb, aww_t, awf, nullptr, (int)BN, 128, 256, 256, 256, 128, 1,
         0,0,0,1,1, 0,0,0,1,1, 0,0,0,1,1, awb, 0, 1.0f);
    k_deform<<<dim3((unsigned)BN), dim3(256), 0, stream>>>(sof, awf, refp, value, t2d);

    // ---- Gate ----
    k_concat<<<dim3((unsigned)((BN * 512 + 255) / 256)), dim3(256), 0, stream>>>(t2f, t2d, gi, BN * 512);
    gemm(true, gi, gw_t, gatev, nullptr, (int)BN, 512, 512, 512, 512, 512, 1,
         0,0,0,1,1, 0,0,0,1,1, 0,0,0,1,1, gb, 2, 1.0f);
    k_ln<<<dim3((unsigned)BN), dim3(256), 0, stream>>>(t2f, t2d, gatev, gng, gnb, t3f, t3b, 1);

    // ---- FFN ----
    gemm(true, t3b, w1_t, nullptr, h1, (int)BN, 1024, 256, 256, 256, 1024, 1,
         0,0,0,1,1, 0,0,0,1,1, 0,0,0,1,1, b1, 1, 1.0f);
    gemm(true, h1, w2_t, ffn, nullptr, (int)BN, 256, 1024, 1024, 1024, 256, 1,
         0,0,0,1,1, 0,0,0,1,1, 0,0,0,1,1, b2, 0, 1.0f);
    k_ln<<<dim3((unsigned)BN), dim3(256), 0, stream>>>(t3f, ffn, nullptr, n3g, n3b, out, nullptr, 2);

    (void)t2b;
}
